// NonLocalTemporalPooling_linear_87840671138504
// MI455X (gfx1250) — matrix operand order repaired and checked by reading, not compiled, not run on hardware
//
#include <hip/hip_runtime.h>
#include <hip/hip_bf16.h>

// ---------------------------------------------------------------------------
// NonLocalTemporalPooling block for MI455X (gfx1250, wave32, WMMA).
// All GEMMs use v_wmma_f32_16x16x32_f16 (f16 inputs, f32 accumulate) with
// 4-wide N-tiling per wave (2-wide in attn_scores; 1 A-tile load amortized over NT WMMAs), branch-free
// inner loop (loads clustered, WMMAs issued back-to-back).
// Softmax stages its row through LDS via global_load_async_to_lds_b128.
// LN / softmax / gelu run in f32; large intermediates in f16 (HBM roofline).
// ---------------------------------------------------------------------------

#define DEVINL __device__ __forceinline__

constexpr int Cc   = 1024;
constexpr int Tt   = 16;
constexpr int Hh   = 14;
constexpr int Ww   = 21;
constexpr int Bb   = 32;
constexpr int HWsz = Hh * Ww;          // 294
constexpr int Ntok = HWsz;             // 294 query tokens per batch
constexpr int Npad = 304;              // 19 * 16 (padded query rows)
constexpr int Mtot = Tt * HWsz;        // 4704 kv tokens per batch (294 * 16)
constexpr int C4   = 4 * Cc;           // 4096
constexpr float LN_EPS = 1e-5f;

typedef __attribute__((ext_vector_type(16))) _Float16 v16h;
typedef __attribute__((ext_vector_type(8)))  _Float16 v8h;
typedef __attribute__((ext_vector_type(8)))  float    v8f;

// ---------------------------------------------------------------------------
// WMMA operand loaders (wave32, 16-bit layouts per CDNA5 ISA 7.12.2)
// A 16x32 (MxK): lane m=lane&15; element e -> K = (e<8?0:16) + (lane>=16?8:0) + (e&7)
// B 32x16 (KxN): lane n=lane&15; element e -> K = (e<8?0:16) + (lane>=16?8:0) + (e&7)
// ---------------------------------------------------------------------------
DEVINL v16h load_a16(const _Float16* __restrict__ p0, int lda, int lane) {
  const _Float16* p = p0 + (size_t)(lane & 15) * lda + ((lane >> 4) << 3);
  v8h lo = *(const v8h*)(p);       // K = k0 .. k0+7
  v8h hi = *(const v8h*)(p + 16);  // K = k0+16 .. k0+23
  v16h r;
#pragma unroll
  for (int i = 0; i < 8; ++i) { r[i] = lo[i]; r[i + 8] = hi[i]; }
  return r;
}

DEVINL v16h load_b16(const _Float16* __restrict__ p0, int ldb, int lane) {
  const _Float16* p = p0 + (size_t)(lane & 15) * ldb + ((lane >> 4) << 3);
  v8h lo = *(const v8h*)(p);       // K = k0 .. k0+7
  v8h hi = *(const v8h*)(p + 16);  // K = k0+16 .. k0+23
  v16h r;
#pragma unroll
  for (int i = 0; i < 8; ++i) { r[i] = lo[i]; r[i + 8] = hi[i]; }
  return r;
}

// K-loop with NT output tiles along N: one A-tile load feeds NT WMMAs.
// Branch-free body: unconditional speculative A prefetch (safe on CDNA5 —
// translation failures are silently dropped), all loads issued before the
// WMMA burst so the scheduler can clause them.
template <int NT>
DEVINL void wmma_loopN(const _Float16* __restrict__ A, int lda,
                       const _Float16* __restrict__ W, int ldb,
                       int K, int lane, v8f (&acc)[NT]) {
#pragma unroll 2
  for (int k = 0; k < K; k += 32) {
    __builtin_prefetch(A + (size_t)(lane & 15) * lda + k + 64, 0, 1);
    v16h a = load_a16(A + k, lda, lane);
    v16h b[NT];
#pragma unroll
    for (int t = 0; t < NT; ++t)
      b[t] = load_b16(W + (size_t)(t * 16) * ldb + k, ldb, lane);
#pragma unroll
    for (int t = 0; t < NT; ++t)
      acc[t] = __builtin_amdgcn_wmma_f32_16x16x32_f16(false, a, false, b[t],
                                                      (short)0, acc[t],
                                                      false, false);
  }
}

// ---------------------------------------------------------------------------
// fp32 -> fp16 conversion (weights)
// ---------------------------------------------------------------------------
__global__ void cvt_f32_f16(const float* __restrict__ src,
                            _Float16* __restrict__ dst, int n) {
  int i = blockIdx.x * blockDim.x + threadIdx.x;
  if (i < n) dst[i] = (_Float16)src[i];
}

// ---------------------------------------------------------------------------
// prep_q: gather last frame (transpose c<->token), +pos2d, LN -> f16 row.
// Also emits x_last (f32) for the residual. Padded rows (n>=294) zeroed.
// ---------------------------------------------------------------------------
__global__ void prep_q_kernel(const float* __restrict__ x_t,
                              const float* __restrict__ pos2d,
                              const float* __restrict__ g,
                              const float* __restrict__ bta,
                              _Float16* __restrict__ q_in,
                              float* __restrict__ x_last) {
  int row = blockIdx.x;                 // b*Npad + n
  int b = row / Npad, n = row % Npad;
  int tid = threadIdx.x;
  if (n >= Ntok) {
    for (int c = tid; c < Cc; c += 256) {
      q_in[(size_t)row * Cc + c] = (_Float16)0.0f;
      x_last[(size_t)row * Cc + c] = 0.0f;
    }
    return;
  }
  float v[4]; float s = 0.f, ss = 0.f;
#pragma unroll
  for (int j = 0; j < 4; ++j) {
    int c = tid * 4 + j;
    float xv = x_t[(((size_t)b * Cc + c) * Tt + (Tt - 1)) * HWsz + n];
    x_last[(size_t)row * Cc + c] = xv;
    float y = xv + pos2d[(size_t)n * Cc + c];
    v[j] = y; s += y; ss += y * y;
  }
  __shared__ float rs[256], rss[256];
  rs[tid] = s; rss[tid] = ss; __syncthreads();
  for (int st = 128; st > 0; st >>= 1) {
    if (tid < st) { rs[tid] += rs[tid + st]; rss[tid] += rss[tid + st]; }
    __syncthreads();
  }
  float mu  = rs[0] * (1.0f / Cc);
  float var = rss[0] * (1.0f / Cc) - mu * mu;
  float inv = rsqrtf(var + LN_EPS);
#pragma unroll
  for (int j = 0; j < 4; ++j) {
    int c = tid * 4 + j;
    q_in[(size_t)row * Cc + c] = (_Float16)((v[j] - mu) * inv * g[c] + bta[c]);
  }
}

// prep_kv: all T frames, +pos3d, LN -> f16 row. One block per kv token.
__global__ void prep_kv_kernel(const float* __restrict__ x_t,
                               const float* __restrict__ pos3d,
                               const float* __restrict__ g,
                               const float* __restrict__ bta,
                               _Float16* __restrict__ kv_in) {
  int row = blockIdx.x;                 // b*Mtot + m
  int b = row / Mtot, m = row % Mtot;
  int tid = threadIdx.x;
  float v[4]; float s = 0.f, ss = 0.f;
#pragma unroll
  for (int j = 0; j < 4; ++j) {
    int c = tid * 4 + j;
    float xv = x_t[((size_t)b * Cc + c) * Mtot + m];
    float y = xv + pos3d[(size_t)m * Cc + c];
    v[j] = y; s += y; ss += y * y;
  }
  __shared__ float rs[256], rss[256];
  rs[tid] = s; rss[tid] = ss; __syncthreads();
  for (int st = 128; st > 0; st >>= 1) {
    if (tid < st) { rs[tid] += rs[tid + st]; rss[tid] += rss[tid + st]; }
    __syncthreads();
  }
  float mu  = rs[0] * (1.0f / Cc);
  float var = rss[0] * (1.0f / Cc) - mu * mu;
  float inv = rsqrtf(var + LN_EPS);
#pragma unroll
  for (int j = 0; j < 4; ++j) {
    int c = tid * 4 + j;
    kv_in[(size_t)row * Cc + c] = (_Float16)((v[j] - mu) * inv * g[c] + bta[c]);
  }
}

// ---------------------------------------------------------------------------
// Projection: Out[m, d] = sum_k A[m,k] * W[d,k]  (f16 out, row-major).
// grid = (Mtiles, Dtiles/4); one wave computes a 16x64 strip (4 WMMA tiles).
// ---------------------------------------------------------------------------
__global__ void __launch_bounds__(32)
gemm_f16_rowout(const _Float16* __restrict__ A, const _Float16* __restrict__ W,
                _Float16* __restrict__ Out, int K, int ldo) {
  int mt = blockIdx.x, nt0 = blockIdx.y * 4, lane = threadIdx.x;
  v8f acc[4] = {};
  wmma_loopN<4>(A + (size_t)mt * 16 * K, K, W + (size_t)nt0 * 16 * K, K, K, lane, acc);
  int m0 = mt * 16 + ((lane >> 4) << 3);
#pragma unroll
  for (int t = 0; t < 4; ++t) {
    int n = (nt0 + t) * 16 + (lane & 15);
#pragma unroll
    for (int r = 0; r < 8; ++r)
      Out[(size_t)(m0 + r) * ldo + n] = (_Float16)acc[t][r];
  }
}

// V projection, stored transposed per batch: Vt[(b*C + d) * Mtot + m].
__global__ void __launch_bounds__(32)
gemm_f16_vtout(const _Float16* __restrict__ A, const _Float16* __restrict__ W,
               _Float16* __restrict__ Vt) {
  int mt = blockIdx.x, nt0 = blockIdx.y * 4, lane = threadIdx.x;
  v8f acc[4] = {};
  wmma_loopN<4>(A + (size_t)mt * 16 * Cc, Cc, W + (size_t)nt0 * 16 * Cc, Cc, Cc, lane, acc);
  int m0g = mt * 16 + ((lane >> 4) << 3);
#pragma unroll
  for (int t = 0; t < 4; ++t) {
    int d = (nt0 + t) * 16 + (lane & 15);
#pragma unroll
    for (int r = 0; r < 8; ++r) {
      int mg = m0g + r;
      int b = mg / Mtot, m = mg % Mtot;
      Vt[((size_t)b * Cc + d) * Mtot + m] = (_Float16)acc[t][r];
    }
  }
}

// S[b, n, m] = (Q . K) / 32 (f32 out). grid = (19, 147, B); 2 m-tiles/wave.
__global__ void __launch_bounds__(32)
attn_scores_kernel(const _Float16* __restrict__ Qh, const _Float16* __restrict__ Kh,
                   float* __restrict__ S) {
  int nt = blockIdx.x, mt0 = blockIdx.y * 2, b = blockIdx.z, lane = threadIdx.x;
  const _Float16* a = Qh + ((size_t)b * Npad + nt * 16) * Cc;
  const _Float16* w = Kh + ((size_t)b * Mtot + mt0 * 16) * Cc;
  v8f acc[2] = {};
  wmma_loopN<2>(a, Cc, w, Cc, Cc, lane, acc);
  const float scale = 0.03125f;  // 1024^-0.5
  int n0 = nt * 16 + ((lane >> 4) << 3);
#pragma unroll
  for (int t = 0; t < 2; ++t) {
    int m = (mt0 + t) * 16 + (lane & 15);
#pragma unroll
    for (int r = 0; r < 8; ++r)
      S[((size_t)b * Npad + n0 + r) * Mtot + m] = acc[t][r] * scale;
  }
}

// ---------------------------------------------------------------------------
// Row softmax over 4704. The row is staged into LDS once with CDNA5 async
// global->LDS copies (ASYNCcnt-tracked), then max/sum/normalize all read LDS.
// ---------------------------------------------------------------------------
__global__ void softmax_kernel(const float* __restrict__ S, _Float16* __restrict__ P) {
  size_t row = blockIdx.x;
  const float* s = S + row * Mtot;
  _Float16*    p = P + row * Mtot;
  int tid = threadIdx.x;
  __shared__ float srow[Mtot];          // 18816 B
  __shared__ float red[256];

  // Async copy: 4704 floats = 1176 x b128 per-lane transfers.
  const unsigned lds0 = (unsigned)(size_t)(const void*)&srow[0];
  for (int j = tid; j < Mtot / 4; j += 256) {
    unsigned     laddr = lds0 + (unsigned)j * 16u;
    const float* gaddr = s + (size_t)j * 4;
    asm volatile("global_load_async_to_lds_b128 %0, %1, off"
                 :: "v"(laddr), "v"(gaddr) : "memory");
  }
  asm volatile("s_wait_asynccnt 0x0" ::: "memory");
  __syncthreads();

  float mx = -1e30f;
  for (int j = tid; j < Mtot; j += 256) mx = fmaxf(mx, srow[j]);
  red[tid] = mx; __syncthreads();
  for (int st = 128; st > 0; st >>= 1) {
    if (tid < st) red[tid] = fmaxf(red[tid], red[tid + st]);
    __syncthreads();
  }
  mx = red[0]; __syncthreads();
  float sum = 0.f;
  for (int j = tid; j < Mtot; j += 256) sum += expf(srow[j] - mx);
  red[tid] = sum; __syncthreads();
  for (int st = 128; st > 0; st >>= 1) {
    if (tid < st) red[tid] += red[tid + st];
    __syncthreads();
  }
  float inv = 1.0f / red[0];
  for (int j = tid; j < Mtot; j += 256)
    p[j] = (_Float16)(expf(srow[j] - mx) * inv);
}

// X[b,n,d] = x_last + sum_m P[b,n,m] * V[b,m,d]. grid = (19, 16, B), 4 d-tiles.
__global__ void __launch_bounds__(32)
attn_out_kernel(const _Float16* __restrict__ P, const _Float16* __restrict__ Vt,
                const float* __restrict__ x_last, float* __restrict__ X) {
  int nt = blockIdx.x, dt0 = blockIdx.y * 4, b = blockIdx.z, lane = threadIdx.x;
  const _Float16* a = P  + ((size_t)b * Npad + nt * 16) * Mtot;
  const _Float16* w = Vt + ((size_t)b * Cc   + dt0 * 16) * Mtot;
  v8f acc[4] = {};
  wmma_loopN<4>(a, Mtot, w, Mtot, Mtot, lane, acc);
  int n0 = nt * 16 + ((lane >> 4) << 3);
#pragma unroll
  for (int t = 0; t < 4; ++t) {
    int d = (dt0 + t) * 16 + (lane & 15);
#pragma unroll
    for (int r = 0; r < 8; ++r) {
      size_t idx = ((size_t)b * Npad + n0 + r) * Cc + d;
      X[idx] = x_last[idx] + acc[t][r];
    }
  }
}

// f32 rows -> LN -> f16 rows (final layernorm before the MLP).
__global__ void ln_f32_f16_kernel(const float* __restrict__ X,
                                  const float* __restrict__ g,
                                  const float* __restrict__ bta,
                                  _Float16* __restrict__ Y) {
  size_t row = blockIdx.x;
  int tid = threadIdx.x;
  float v[4]; float s = 0.f, ss = 0.f;
#pragma unroll
  for (int j = 0; j < 4; ++j) {
    int c = tid * 4 + j;
    float y = X[row * Cc + c];
    v[j] = y; s += y; ss += y * y;
  }
  __shared__ float rs[256], rss[256];
  rs[tid] = s; rss[tid] = ss; __syncthreads();
  for (int st = 128; st > 0; st >>= 1) {
    if (tid < st) { rs[tid] += rs[tid + st]; rss[tid] += rss[tid + st]; }
    __syncthreads();
  }
  float mu  = rs[0] * (1.0f / Cc);
  float var = rss[0] * (1.0f / Cc) - mu * mu;
  float inv = rsqrtf(var + LN_EPS);
#pragma unroll
  for (int j = 0; j < 4; ++j) {
    int c = tid * 4 + j;
    Y[row * Cc + c] = (_Float16)((v[j] - mu) * inv * g[c] + bta[c]);
  }
}

// G = gelu(H @ w1^T + b1), exact gelu. grid = (608, 64), 4 n-tiles per wave.
__global__ void __launch_bounds__(32)
mlp1_kernel(const _Float16* __restrict__ Hn, const _Float16* __restrict__ W1,
            const float* __restrict__ b1, _Float16* __restrict__ G) {
  int mt = blockIdx.x, nt0 = blockIdx.y * 4, lane = threadIdx.x;
  v8f acc[4] = {};
  wmma_loopN<4>(Hn + (size_t)mt * 16 * Cc, Cc, W1 + (size_t)nt0 * 16 * Cc, Cc, Cc, lane, acc);
  int m0 = mt * 16 + ((lane >> 4) << 3);
#pragma unroll
  for (int t = 0; t < 4; ++t) {
    int col = (nt0 + t) * 16 + (lane & 15);
    float bb = b1[col];
#pragma unroll
    for (int r = 0; r < 8; ++r) {
      float v = acc[t][r] + bb;
      float ge = 0.5f * v * (1.0f + erff(v * 0.70710678f));
      G[(size_t)(m0 + r) * C4 + col] = (_Float16)ge;
    }
  }
}

// out[b,c,h,w] = X[b,n,c] + G @ w2^T + b2  (scatter into channel-first layout)
__global__ void __launch_bounds__(32)
mlp2_kernel(const _Float16* __restrict__ G, const _Float16* __restrict__ W2,
            const float* __restrict__ b2, const float* __restrict__ X,
            float* __restrict__ out) {
  int mt = blockIdx.x, ct0 = blockIdx.y * 4, lane = threadIdx.x;
  v8f acc[4] = {};
  wmma_loopN<4>(G + (size_t)mt * 16 * C4, C4, W2 + (size_t)ct0 * 16 * C4, C4, C4, lane, acc);
  int m0 = mt * 16 + ((lane >> 4) << 3);
#pragma unroll
  for (int t = 0; t < 4; ++t) {
    int c = (ct0 + t) * 16 + (lane & 15);
    float bb = b2[c];
#pragma unroll
    for (int r = 0; r < 8; ++r) {
      int rowg = m0 + r;
      int b = rowg / Npad, n = rowg % Npad;
      if (n < Ntok)
        out[((size_t)b * Cc + c) * HWsz + n] =
            acc[t][r] + bb + X[(size_t)rowg * Cc + c];
    }
  }
}

// ---------------------------------------------------------------------------
extern "C" void kernel_launch(void* const* d_in, const int* in_sizes, int n_in,
                              void* d_out, int out_size, void* d_ws, size_t ws_size,
                              hipStream_t stream) {
  const float* x_t   = (const float*)d_in[0];
  const float* ln2_g = (const float*)d_in[1];
  const float* ln2_b = (const float*)d_in[2];
  const float* ln3_g = (const float*)d_in[3];
  const float* ln3_b = (const float*)d_in[4];
  const float* lnf_g = (const float*)d_in[5];
  const float* lnf_b = (const float*)d_in[6];
  const float* pos2d = (const float*)d_in[7];
  const float* pos3d = (const float*)d_in[8];
  const float* wq    = (const float*)d_in[9];
  const float* wk    = (const float*)d_in[10];
  const float* wv    = (const float*)d_in[11];
  const float* w1    = (const float*)d_in[12];
  const float* b1    = (const float*)d_in[13];
  const float* w2    = (const float*)d_in[14];
  const float* b2    = (const float*)d_in[15];
  float* out = (float*)d_out;

  // -------- workspace carve-up (with lifetime-based aliasing) --------------
  char* ws = (char*)d_ws;
  size_t off = 0;
  auto carve = [&](size_t bytes) -> char* {
    char* p = ws + off;
    off += (bytes + 255) & ~(size_t)255;
    return p;
  };
  const size_t szW    = (size_t)Cc * Cc * sizeof(_Float16);
  const size_t szW14  = (size_t)C4 * Cc * sizeof(_Float16);
  const size_t szQin  = (size_t)Bb * Npad * Cc * sizeof(_Float16);
  const size_t szXf   = (size_t)Bb * Npad * Cc * sizeof(float);
  const size_t szKVi  = (size_t)Bb * Mtot * Cc * sizeof(_Float16);
  const size_t szS    = (size_t)Bb * Npad * Mtot * sizeof(float);
  const size_t szP    = (size_t)Bb * Npad * Mtot * sizeof(_Float16);
  const size_t szG    = (size_t)Bb * Npad * C4 * sizeof(_Float16);

  _Float16* wq_h = (_Float16*)carve(szW);
  _Float16* wk_h = (_Float16*)carve(szW);
  _Float16* wv_h = (_Float16*)carve(szW);
  _Float16* w1_h = (_Float16*)carve(szW14);
  _Float16* w2_h = (_Float16*)carve(szW14);
  _Float16* q_in_h = (_Float16*)carve(szQin);     // later reused as LN(x) f16
  float*    x_last = (float*)carve(szXf);
  _Float16* Q_h    = (_Float16*)carve(szQin);
  float*    x_f32  = (float*)carve(szXf);
  // kv region: kv_in lives until V gemm; then S (+P) reuse the same space.
  size_t kvRegion = szKVi > (szS + szP + 512) ? szKVi : (szS + szP + 512);
  char* kvBase = carve(kvRegion);
  _Float16* kv_in_h = (_Float16*)kvBase;
  float*    S_f32   = (float*)kvBase;
  _Float16* P_h     = (_Float16*)(kvBase + ((szS + 255) & ~(size_t)255));
  // K region: K_h lives until scores; then gelu activations reuse it.
  size_t kRegion = szKVi > szG ? szKVi : szG;
  char* kBase = carve(kRegion);
  _Float16* K_h = (_Float16*)kBase;
  _Float16* G_h = (_Float16*)kBase;
  _Float16* Vt_h = (_Float16*)carve(szKVi);
  _Float16* H_h  = q_in_h;   // alias: q_in dead after Q projection

  // -------- pipeline -------------------------------------------------------
  // 1) weights -> f16
  {
    int n = Cc * Cc;
    cvt_f32_f16<<<(n + 255) / 256, 256, 0, stream>>>(wq, wq_h, n);
    cvt_f32_f16<<<(n + 255) / 256, 256, 0, stream>>>(wk, wk_h, n);
    cvt_f32_f16<<<(n + 255) / 256, 256, 0, stream>>>(wv, wv_h, n);
    int n4 = C4 * Cc;
    cvt_f32_f16<<<(n4 + 255) / 256, 256, 0, stream>>>(w1, w1_h, n4);
    cvt_f32_f16<<<(n4 + 255) / 256, 256, 0, stream>>>(w2, w2_h, n4);
  }
  // 2) transpose + pos + LN prep
  prep_q_kernel<<<Bb * Npad, 256, 0, stream>>>(x_t, pos2d, ln2_g, ln2_b, q_in_h, x_last);
  prep_kv_kernel<<<Bb * Mtot, 256, 0, stream>>>(x_t, pos3d, ln3_g, ln3_b, kv_in_h);
  // 3) projections (WMMA f16, 16x64 strip per wave)
  gemm_f16_rowout<<<dim3(Bb * Npad / 16, Cc / 64), 32, 0, stream>>>(q_in_h, wq_h, Q_h, Cc, Cc);
  gemm_f16_rowout<<<dim3(Bb * Mtot / 16, Cc / 64), 32, 0, stream>>>(kv_in_h, wk_h, K_h, Cc, Cc);
  gemm_f16_vtout <<<dim3(Bb * Mtot / 16, Cc / 64), 32, 0, stream>>>(kv_in_h, wv_h, Vt_h);
  // 4) attention
  attn_scores_kernel<<<dim3(Npad / 16, Mtot / 32, Bb), 32, 0, stream>>>(Q_h, K_h, S_f32);
  softmax_kernel<<<Bb * Npad, 256, 0, stream>>>(S_f32, P_h);
  attn_out_kernel<<<dim3(Npad / 16, Cc / 64, Bb), 32, 0, stream>>>(P_h, Vt_h, x_last, x_f32);
  // 5) MLP with residual, scattered into (B, C, H, W) output
  ln_f32_f16_kernel<<<Bb * Npad, 256, 0, stream>>>(x_f32, lnf_g, lnf_b, H_h);
  mlp1_kernel<<<dim3(Bb * Npad / 16, C4 / 64), 32, 0, stream>>>(H_h, w1_h, b1, G_h);
  mlp2_kernel<<<dim3(Bb * Npad / 16, Cc / 64), 32, 0, stream>>>(G_h, w2_h, b2, x_f32, out);
}
